// QuesAnsModel_41858751266982
// MI455X (gfx1250) — compile-verified, matrix operand order checked by reading, not run on hardware
//
#include <hip/hip_runtime.h>

// ---------------- problem constants ----------------
#define V_DIM 100000
#define M_DIM 200
#define D_DIM 128
#define MPAD  208                    // 13 * 16 (M padded to tile multiple)
#define MAT_ELEMS (MPAD * D_DIM)     // 26624
#define KSTEPS 3125                  // V_DIM / 32
#define KSPLIT 125                   // workgroups in GEMM grid
#define KSTEPS_PER_WG (KSTEPS / KSPLIT)  // 25
#define GEMM_THREADS 832             // 26 waves: 13 mtiles x 2 half-strips
#define A_ROW_PITCH 36               // 32 k-floats + 4 pad (bank-conflict-free)

typedef __attribute__((ext_vector_type(16))) __bf16          v16bf;
typedef __attribute__((ext_vector_type(2)))  __bf16          v2bf;
typedef __attribute__((ext_vector_type(8)))  unsigned int    v8ui;
typedef __attribute__((ext_vector_type(8)))  float           v8f;

__device__ __forceinline__ unsigned short bf16_bits(float f) {
  unsigned int u = __float_as_uint(f);
  u += 0x7FFFu + ((u >> 16) & 1u);   // round-to-nearest-even
  return (unsigned short)(u >> 16);
}

// pack two f32 into a dword of 2 bf16 (lo = a, hi = b), RNE
__device__ __forceinline__ unsigned int pack2_bf16(float a, float b) {
#if __has_builtin(__builtin_amdgcn_cvt_pk_bf16_f32)
  v2bf r = __builtin_amdgcn_cvt_pk_bf16_f32(a, b);
  return __builtin_bit_cast(unsigned int, r);
#else
  return (unsigned int)bf16_bits(a) | ((unsigned int)bf16_bits(b) << 16);
#endif
}

// async 16B copy: global (saddr + 32-bit voffset) -> LDS byte offset (ASYNCcnt)
__device__ __forceinline__ void async_copy16(unsigned ldsOff, const float* gbase,
                                             unsigned gOff) {
  asm volatile("global_load_async_to_lds_b128 %0, %1, %2"
               :: "v"(ldsOff), "v"(gOff), "s"(gbase) : "memory");
}

__device__ __forceinline__ void wait_async0() {
#if __has_builtin(__builtin_amdgcn_s_wait_asynccnt)
  __builtin_amdgcn_s_wait_asynccnt(0);
#else
  asm volatile("s_wait_asynccnt 0x0" ::: "memory");
#endif
}

// ---------------- phase 1: ques_d = question @ Wb.T  [1,128] ----------------
__global__ __launch_bounds__(256)
void k_quesd(const float* __restrict__ q, const float* __restrict__ Wb,
             float* __restrict__ qd) {
  __shared__ float red[256];
  const int d = blockIdx.x;
  const int tid = threadIdx.x;
  const float* row = Wb + (size_t)d * V_DIM;
  float s = 0.f;
  for (int v = tid; v < V_DIM; v += 256) s += q[v] * row[v];
  red[tid] = s;
  __syncthreads();
  for (int off = 128; off > 0; off >>= 1) {
    if (tid < off) red[tid] += red[tid + off];
    __syncthreads();
  }
  if (tid == 0) qd[d] = red[0];
}

// ---------------- phase 2/3: fused bf16-WMMA GEMM for current_A/current_C ----
// partials[wg][mat][MPAD][128], mat 0 = A (memory@Wa.T), mat 1 = C (memory@Wc.T)
__global__ __launch_bounds__(GEMM_THREADS)
void k_gemm_ac(const float* __restrict__ mem,
               const float* __restrict__ Wa,
               const float* __restrict__ Wc,
               float* __restrict__ partials) {
  // B slabs staged in WMMA-fragment order: [buf][mat][ntile][lane][half-idx]
  __shared__ __attribute__((aligned(32))) unsigned short ldsB[2][2][8][32][16];
  // A slab (raw f32, rows 0..199 of this k-step), filled by async-to-LDS DMA
  __shared__ __attribute__((aligned(16))) float ldsA[2][M_DIM][A_ROW_PITCH];

  const int tid   = threadIdx.x;
  const int wg    = blockIdx.x;
  const int lane  = tid & 31;
  const int wave  = tid >> 5;        // 0..25
  const int mtile = wave >> 1;       // 0..12
  const int half  = wave & 1;        // 0: ntiles 0..3, 1: ntiles 4..7
  const int tbase = half * 4;
  const int khalf = lane >> 4;       // A layout: lanes 16..31 hold K+8 group
  const int row   = mtile * 16 + (lane & 15);

  const int ksBegin = wg * KSTEPS_PER_WG;

  // Issue async global->LDS copies for the A slab of one k-step.
  // 200 rows x 32 f32 = 1600 16B chunks, coalesced 128B per 8 consecutive tids.
  auto stageA = [&](int buf, int k0) {
    const unsigned ldsBase = (unsigned)(size_t)(&ldsA[buf][0][0]);
    for (int c = tid; c < M_DIM * 8; c += GEMM_THREADS) {
      const int arow = c >> 3;
      const int seg  = c & 7;
      const unsigned gOff   = (unsigned)((arow * V_DIM + k0 + seg * 4) * 4);
      const unsigned ldsOff = ldsBase + (unsigned)((arow * A_ROW_PITCH + seg * 4) * 4);
      async_copy16(ldsOff, mem, gOff);
    }
  };

  // Stage one k-slab of Wa/Wc in fragment order.
  // 1024 chunks of 8 consecutive-K elements; each chunk is 32 contiguous
  // global bytes -> one 16B ds_store_b128 at its fragment position.
  auto stageB = [&](int buf, int k0) {
    for (int c = tid; c < 1024; c += GEMM_THREADS) {
      const int mat = c >> 9;          // 0 = Wa, 1 = Wc
      const int rem = c & 511;
      const int n   = rem >> 2;        // 0..127 (row of W == output column)
      const int r   = rem & 3;         // k-run: kbase = 8*(r&1) + 16*(r>>1)
      const int s   = r & 1;
      const int hi  = r >> 1;
      const float* src = (mat ? Wc : Wa) + (size_t)n * V_DIM + k0 + 8 * s + 16 * hi;
      const float4 f0 = *(const float4*)(src);
      const float4 f1 = *(const float4*)(src + 4);
      uint4 pk;
      pk.x = pack2_bf16(f0.x, f0.y);
      pk.y = pack2_bf16(f0.z, f0.w);
      pk.z = pack2_bf16(f1.x, f1.y);
      pk.w = pack2_bf16(f1.z, f1.w);
      *(uint4*)(&ldsB[buf][mat][n >> 4][(n & 15) + 16 * s][8 * hi]) = pk;
    }
  };

  const v8f vzero = {0.f, 0.f, 0.f, 0.f, 0.f, 0.f, 0.f, 0.f};
  v8f accA[4], accC[4];
#pragma unroll
  for (int t = 0; t < 4; ++t) { accA[t] = vzero; accC[t] = vzero; }

  stageA(0, ksBegin * 32);
  stageB(0, ksBegin * 32);
  wait_async0();
  __syncthreads();

  int buf = 0;
  for (int i = 0; i < KSTEPS_PER_WG; ++i) {
    const int k0 = (ksBegin + i) * 32;
    if (i + 1 < KSTEPS_PER_WG) {
      stageA(buf ^ 1, k0 + 32);      // fire-and-forget DMA, overlaps compute
      stageB(buf ^ 1, k0 + 32);
    }

    // A fragment from LDS slab (f32 -> packed bf16). 16-bit A 16x32 layout:
    //   halves 0..7  -> K = 8*khalf + h ; halves 8..15 -> K = 16 + 8*khalf + (h-8)
    v8ui au;
    if (row < M_DIM) {
      const float* ap = &ldsA[buf][row][khalf * 8];
      const float4 a0 = *(const float4*)(ap);
      const float4 a1 = *(const float4*)(ap + 4);
      const float4 a2 = *(const float4*)(ap + 16);
      const float4 a3 = *(const float4*)(ap + 20);
      au[0] = pack2_bf16(a0.x, a0.y); au[1] = pack2_bf16(a0.z, a0.w);
      au[2] = pack2_bf16(a1.x, a1.y); au[3] = pack2_bf16(a1.z, a1.w);
      au[4] = pack2_bf16(a2.x, a2.y); au[5] = pack2_bf16(a2.z, a2.w);
      au[6] = pack2_bf16(a3.x, a3.y); au[7] = pack2_bf16(a3.z, a3.w);
    } else {
#pragma unroll
      for (int h = 0; h < 8; ++h) au[h] = 0u;   // zero-pad rows 200..207
    }
    const v16bf afrag = __builtin_bit_cast(v16bf, au);

    // Load all 8 B fragments, then 8 back-to-back WMMAs.
    v16bf bA[4], bC[4];
#pragma unroll
    for (int tt = 0; tt < 4; ++tt)
      bA[tt] = *(const v16bf*)(&ldsB[buf][0][tbase + tt][lane][0]);
#pragma unroll
    for (int tt = 0; tt < 4; ++tt)
      bC[tt] = *(const v16bf*)(&ldsB[buf][1][tbase + tt][lane][0]);
#pragma unroll
    for (int tt = 0; tt < 4; ++tt)
      accA[tt] = __builtin_amdgcn_wmma_f32_16x16x32_bf16(
          false, afrag, false, bA[tt], (short)0, accA[tt], false, false);
#pragma unroll
    for (int tt = 0; tt < 4; ++tt)
      accC[tt] = __builtin_amdgcn_wmma_f32_16x16x32_bf16(
          false, afrag, false, bC[tt], (short)0, accC[tt], false, false);

    wait_async0();                   // own DMA for next buffer complete
    __syncthreads();                 // whole workgroup's staging visible
    buf ^= 1;
  }

  // epilogue: write partial tiles (f32 16x16 D layout: VGPR r -> M=r+8*(lane>>4), N=lane&15)
  float* pA = partials + (size_t)wg * 2 * MAT_ELEMS;
  float* pC = pA + MAT_ELEMS;
#pragma unroll
  for (int tt = 0; tt < 4; ++tt) {
    const int n = (tbase + tt) * 16 + (lane & 15);
#pragma unroll
    for (int r = 0; r < 8; ++r) {
      const int m = mtile * 16 + r + khalf * 8;
      pA[m * D_DIM + n] = accA[tt][r];
      pC[m * D_DIM + n] = accC[tt][r];
    }
  }
}

// ---------------- reduce partials + add temporal ----------------
__global__ __launch_bounds__(256)
void k_reduce(const float* __restrict__ part,
              const float* __restrict__ tA, const float* __restrict__ tC,
              float* __restrict__ cA, float* __restrict__ cC) {
  const int idx = blockIdx.x * 256 + threadIdx.x;
  if (idx >= 2 * MAT_ELEMS) return;
  const int mat = idx / MAT_ELEMS;
  const int r   = idx - mat * MAT_ELEMS;
  const int m   = r >> 7;
  const int n   = r & 127;
  float s = 0.f;
  if (m < M_DIM) s = mat ? tC[m * D_DIM + n] : tA[m * D_DIM + n];
  for (int w = 0; w < KSPLIT; ++w)
    s += part[((size_t)w * 2 + mat) * MAT_ELEMS + r];
  (mat ? cC : cA)[r] = s;
}

// ---------------- 3 memory hops (tiny, single block) ----------------
__global__ __launch_bounds__(256)
void k_hops(const float* __restrict__ cA, const float* __restrict__ cC,
            const float* __restrict__ qd, float* __restrict__ o_out) {
  __shared__ float sQ[D_DIM];
  __shared__ float sP[M_DIM];
  __shared__ float sMS[2];
  const int tid = threadIdx.x;
  if (tid < D_DIM) sQ[tid] = qd[tid];
  __syncthreads();
  for (int hop = 0; hop < 3; ++hop) {
    if (tid < M_DIM) {
      const float* rowp = cA + (size_t)tid * D_DIM;
      float a = 0.f;
      for (int d = 0; d < D_DIM; ++d) a += sQ[d] * rowp[d];
      sP[tid] = a;
    }
    __syncthreads();
    if (tid == 0) {
      float mx = -3.4e38f;
      for (int m = 0; m < M_DIM; ++m) mx = fmaxf(mx, sP[m]);
      sMS[0] = mx;
    }
    __syncthreads();
    if (tid < M_DIM) sP[tid] = __expf(sP[tid] - sMS[0]);
    __syncthreads();
    if (tid == 0) {
      float s = 0.f;
      for (int m = 0; m < M_DIM; ++m) s += sP[m];
      sMS[1] = s;
    }
    __syncthreads();
    float newo = 0.f;
    if (tid < D_DIM) {
      float acc = 0.f;
      for (int m = 0; m < M_DIM; ++m) acc += sP[m] * cC[(size_t)m * D_DIM + tid];
      newo = acc / sMS[1] + sQ[tid];
    }
    __syncthreads();
    if (tid < D_DIM) sQ[tid] = newo;
    __syncthreads();
  }
  if (tid < D_DIM) o_out[tid] = sQ[tid];
}

// ---------------- phase 5: out = o @ Wout.T  [1,V] ----------------
__global__ __launch_bounds__(256)
void k_out(const float* __restrict__ o, const float* __restrict__ Wout,
           float* __restrict__ out) {
  __shared__ float sO[D_DIM];
  if (threadIdx.x < D_DIM) sO[threadIdx.x] = o[threadIdx.x];
  __syncthreads();
  const int v = blockIdx.x * 256 + threadIdx.x;
  if (v >= V_DIM) return;
  const float4* rowp = (const float4*)(Wout + (size_t)v * D_DIM);
  float acc = 0.f;
#pragma unroll 8
  for (int i = 0; i < D_DIM / 4; ++i) {
    const float4 w = rowp[i];
    acc += w.x * sO[4 * i + 0] + w.y * sO[4 * i + 1] +
           w.z * sO[4 * i + 2] + w.w * sO[4 * i + 3];
  }
  out[v] = acc;
}

// ---------------- launcher ----------------
extern "C" void kernel_launch(void* const* d_in, const int* in_sizes, int n_in,
                              void* d_out, int out_size, void* d_ws, size_t ws_size,
                              hipStream_t stream) {
  const float* question = (const float*)d_in[0];
  const float* memory   = (const float*)d_in[1];
  const float* Wa       = (const float*)d_in[2];
  const float* Wb       = (const float*)d_in[3];
  const float* Wc       = (const float*)d_in[4];
  const float* Wout     = (const float*)d_in[5];
  const float* tA       = (const float*)d_in[6];
  const float* tC       = (const float*)d_in[7];
  float* out = (float*)d_out;

  // workspace layout (floats): qd[128] | o[128] | cA[26624] | cC[26624] | partials[125*2*26624]
  float* ws = (float*)d_ws;
  float* qd = ws;
  float* o  = ws + 128;
  float* cA = ws + 256;
  float* cC = cA + MAT_ELEMS;
  float* partials = cC + MAT_ELEMS;

  k_quesd  <<<dim3(D_DIM),                 dim3(256),          0, stream>>>(question, Wb, qd);
  k_gemm_ac<<<dim3(KSPLIT),                dim3(GEMM_THREADS), 0, stream>>>(memory, Wa, Wc, partials);
  k_reduce <<<dim3((2 * MAT_ELEMS) / 256), dim3(256),          0, stream>>>(partials, tA, tC, cA, cC);
  k_hops   <<<dim3(1),                     dim3(256),          0, stream>>>(cA, cC, qd, o);
  k_out    <<<dim3((V_DIM + 255) / 256),   dim3(256),          0, stream>>>(o, Wout, out);
}
